// CrossWinAttention_63410897158937
// MI455X (gfx1250) — matrix operand order repaired and checked by reading, not compiled, not run on hardware
//
#include <hip/hip_runtime.h>
#include <hip/hip_bf16.h>

// ---------------- problem constants ----------------
#define L_WIN   256      // X*Y windows
#define T_TOK   384      // N*W1*W2 tokens per window
#define DIM     128
#define HEADS   4
#define DH      32
#define NREP    6        // mean over n
#define ROWS_OUT 64      // W1*W2 rows after mean
// strides in the (b,n,x,y,w1,w2,d) input layout
#define N_STRIDE  2097152ull   // X*Y*W1*W2*DIM
#define L_STRIDE  8192ull      // W1*W2*DIM

typedef __attribute__((ext_vector_type(16))) _Float16 v16h;
typedef __attribute__((ext_vector_type(8)))  _Float16 v8h;
typedef __attribute__((ext_vector_type(8)))  float    v8f;

// ---- WMMA f32_16x16x32_f16 fragment index maps (wave32) ----
// A (16x32, f16): lane -> row = lane&15 ; element e -> K = (lane>>4)*16 + e (this code's k order, not the ISA table's)
// B (32x16, f16): lane -> col = lane&15 ; element e -> K = (lane>>4)*16 + e (this code's k order, matching A)
// C (16x16, f32): lane -> col = lane&15 ; element r -> row = r + (lane>>4)*8

__device__ __forceinline__ int a_kidx(int e, int hi) { return e + hi * 16; }

__device__ __forceinline__ v16h cat8(v8h a, v8h b) {
    return __builtin_shufflevector(a, b, 0,1,2,3,4,5,6,7,8,9,10,11,12,13,14,15);
}
// A fragment: one contiguous 32B chunk (K = hi*16 .. hi*16+15) of the (row, 32-wide K-block) strip
__device__ __forceinline__ v16h loadA(const _Float16* rowK0, int hi) {
    v8h lo = *(const v8h*)(rowK0 + hi * 16);
    v8h hh = *(const v8h*)(rowK0 + hi * 16 + 8);
    return cat8(lo, hh);
}

// stage a 128x128 f32 weight into LDS as f16, pre-swizzled into B-fragment order:
// W[k][c] -> sW[(nt*8 + kb)*256 + n*16 + e]  with kb=k/16, e=k%16, nt=c/16, n=c%16
__device__ __forceinline__ void stage_weight(const float* __restrict__ W,
                                             _Float16* sW, int tid) {
    for (int i = tid; i < DIM * DIM; i += 256) {
        const int kk = i >> 7, c = i & 127;
        sW[(((c >> 4) * 8 + (kk >> 4)) << 8) + ((c & 15) << 4) + (kk & 15)] =
            (_Float16)W[i];
    }
}

// =====================================================================
// layernorm + projection (384x128 @ 128x128), templated on V-transpose
// =====================================================================
template<bool VT>
__device__ __forceinline__ void ln_project(const float* __restrict__ src,
                                           const float* __restrict__ gam,
                                           const float* __restrict__ bet,
                                           const float* __restrict__ W,
                                           const float* __restrict__ bias,
                                           _Float16* dst, _Float16* sW,
                                           int l, int wave, int m, int hi, int tid)
{
    stage_weight(W, sW, tid);
    __syncthreads();

    for (int it = 0; it < 3; ++it) {             // this wave's 3 M-tiles
        const int mt  = wave + it * 8;
        const int t   = mt * 16 + m;
        const int nn  = t >> 6;
        const int loc = t & 63;
        const float* row = src + (size_t)nn * N_STRIDE + (size_t)l * L_STRIDE
                         + (size_t)loc * DIM;

        // lane loads its 64 K-values (partner lane ^16 holds the other 64)
        float xv[64];
        float s = 0.f, ss = 0.f;
#pragma unroll
        for (int f = 0; f < 4; ++f)
#pragma unroll
            for (int half = 0; half < 2; ++half) {
                const int kb = f * 32 + hi * 16 + half * 8;
#pragma unroll
                for (int j = 0; j < 8; ++j) {
                    float xx = row[kb + j];
                    xv[f * 16 + half * 8 + j] = xx;
                    s += xx; ss += xx * xx;
                }
            }
        s  += __shfl_xor(s, 16, 32);
        ss += __shfl_xor(ss, 16, 32);
        const float mean = s * (1.f / 128.f);
        const float var  = ss * (1.f / 128.f) - mean * mean;
        const float rstd = rsqrtf(var + 1e-5f);

        v16h A[4];
#pragma unroll
        for (int f = 0; f < 4; ++f)
#pragma unroll
            for (int e = 0; e < 16; ++e) {
                const int kk = f * 32 + a_kidx(e, hi);
                A[f][e] = (_Float16)((xv[f * 16 + e] - mean) * rstd * gam[kk] + bet[kk]);
            }

        // K-outermost with 8 live N-tile accumulators: consecutive WMMAs are
        // independent (different C), so the f16-WMMA RAW hazard is hidden.
        v8f c[8];
#pragma unroll
        for (int nt = 0; nt < 8; ++nt) c[nt] = (v8f){};
#pragma unroll
        for (int f = 0; f < 4; ++f)
#pragma unroll
            for (int nt = 0; nt < 8; ++nt) {
                const v16h B =
                    *(const v16h*)(sW + ((nt * 8 + f * 2 + hi) << 8) + (m << 4));
                c[nt] = __builtin_amdgcn_wmma_f32_16x16x32_f16(
                            false, A[f], false, B, (short)0, c[nt], false, false);
            }

#pragma unroll
        for (int nt = 0; nt < 8; ++nt) {
            const float bb = bias[nt * 16 + m];
            if constexpr (VT) {
                // transposed store: 8 consecutive tokens -> one 16B ds store
                v8h pk;
#pragma unroll
                for (int r = 0; r < 8; ++r) pk[r] = (_Float16)(c[nt][r] + bb);
                *(v8h*)(dst + (nt * 16 + m) * T_TOK + mt * 16 + hi * 8) = pk;
            } else {
#pragma unroll
                for (int r = 0; r < 8; ++r)
                    dst[(mt * 16 + r + hi * 8) * DIM + nt * 16 + m] =
                        (_Float16)(c[nt][r] + bb);
            }
        }
    }
    __syncthreads();
}

// =====================================================================
// fused cross-window attention kernel: one workgroup per window
// =====================================================================
extern "C" __global__ void __launch_bounds__(256, 1)
xwin_attn_kernel(const float* __restrict__ q, const float* __restrict__ k,
                 const float* __restrict__ v,
                 const float* __restrict__ lqg, const float* __restrict__ lqb,
                 const float* __restrict__ lkg, const float* __restrict__ lkb,
                 const float* __restrict__ lvg, const float* __restrict__ lvb,
                 const float* __restrict__ Wq, const float* __restrict__ bq,
                 const float* __restrict__ Wk, const float* __restrict__ bk,
                 const float* __restrict__ Wv, const float* __restrict__ bv,
                 const float* __restrict__ Wp, const float* __restrict__ bp,
                 float* __restrict__ out)
{
    extern __shared__ char smem_raw[];
    _Float16* sQ  = (_Float16*)smem_raw;      // 384x128 token-major (96KB) -> attn out -> Abar
    _Float16* sK  = sQ + T_TOK * DIM;         // 384x128 token-major (96KB)
    _Float16* sVt = sK + T_TOK * DIM;         // 128x384 dh-major (TRANSPOSED V, 96KB)
    _Float16* sW  = sVt + T_TOK * DIM;        // 32KB: swizzled weight OR softmax scratch

    const int l    = blockIdx.x;
    const int tid  = threadIdx.x;
    const int wave = tid >> 5;
    const int lane = tid & 31;
    const int m    = lane & 15;   // fragment row (A) / col (B,C)
    const int hi   = lane >> 4;

    // Phase 1: layernorm + projections (specialized, no inner branches)
    ln_project<false>(q, lqg, lqb, Wq, bq, sQ,  sW, l, wave, m, hi, tid);
    ln_project<false>(k, lkg, lkb, Wk, bk, sK,  sW, l, wave, m, hi, tid);
    ln_project<true >(v, lvg, lvb, Wv, bv, sVt, sW, l, wave, m, hi, tid);

    // =================================================================
    // Phase 2: flash attention per (head, q-tile); 96 tasks over 8 waves
    //   - 1/sqrt(dh) folded into the Q fragment (once per task)
    //   - half-tile running max (one per 8-row lane half) instead of per-row max
    //   - softmax denominators accumulated with an extra WMMA vs ones
    //   - attn output overwrites the exact sQ slice it consumed
    // =================================================================
    {
        _Float16* pscr = sW + wave * (16 * 32);     // per-wave P bounce (1KB)
        v16h onesB;
#pragma unroll
        for (int e = 0; e < 16; ++e) onesB[e] = (_Float16)1.0f;

        for (int task = wave; task < HEADS * (T_TOK / 16); task += 8) {
            const int h  = task / (T_TOK / 16);
            const int mt = task % (T_TOK / 16);
            const int co = h * DH;

            v16h aQ = loadA(sQ + (mt * 16 + m) * DIM + co, hi);
#pragma unroll
            for (int e = 0; e < 16; ++e)        // fold in 1/sqrt(DH)
                aQ[e] = (_Float16)((float)aQ[e] * 0.17677669529663687f);

            float mtot = -1e30f;
            v8f acc0 = {}, acc1 = {}, accL = {};

            for (int c = 0; c < T_TOK / 32; ++c) {       // 12 chunks of 32 kv tokens
                const int t0 = c * 32;
                // K fragments: dh-contiguous per lane -> single 32B read each
                const v16h bK0 = *(const v16h*)(sK + (t0 + m) * DIM + co + hi * 16);
                const v16h bK1 = *(const v16h*)(sK + (t0 + 16 + m) * DIM + co + hi * 16);
                const v8f z = {};
                v8f s0 = __builtin_amdgcn_wmma_f32_16x16x32_f16(
                             false, aQ, false, bK0, (short)0, z, false, false);
                v8f s1 = __builtin_amdgcn_wmma_f32_16x16x32_f16(
                             false, aQ, false, bK1, (short)0, z, false, false);

                // half-tile max (8 rows per lane half): in-lane reduce, then 4 cross-lane xors
                float tmx = fmaxf(s0[0], s1[0]);
#pragma unroll
                for (int r = 1; r < 8; ++r) tmx = fmaxf(tmx, fmaxf(s0[r], s1[r]));
                tmx = fmaxf(tmx, __shfl_xor(tmx, 1, 32));
                tmx = fmaxf(tmx, __shfl_xor(tmx, 2, 32));
                tmx = fmaxf(tmx, __shfl_xor(tmx, 4, 32));
                tmx = fmaxf(tmx, __shfl_xor(tmx, 8, 32));
                const float mn    = fmaxf(mtot, tmx);
                const float alpha = __expf(mtot - mn);
                mtot = mn;

                // p = exp(s - mn), bounce through per-wave LDS (C-frag -> A-frag)
#pragma unroll
                for (int r = 0; r < 8; ++r) {
                    const int rowi = r + hi * 8;
                    pscr[rowi * 32 + m]      = (_Float16)__expf(s0[r] - mn);
                    pscr[rowi * 32 + 16 + m] = (_Float16)__expf(s1[r] - mn);
                }
                // rescale accumulators while the stores land
#pragma unroll
                for (int r = 0; r < 8; ++r) {
                    acc0[r] *= alpha; acc1[r] *= alpha; accL[r] *= alpha;
                }
                asm volatile("s_wait_dscnt 0" ::: "memory");
                const v16h pA = loadA(pscr + m * 32, hi);
                asm volatile("s_wait_dscnt 0" ::: "memory");

                // V fragments from transposed V: token-contiguous -> 32B reads
                const v16h bV0 = *(const v16h*)(sVt + (co + m) * T_TOK + t0 + hi * 16);
                const v16h bV1 = *(const v16h*)(sVt + (co + 16 + m) * T_TOK + t0 + hi * 16);
                acc0 = __builtin_amdgcn_wmma_f32_16x16x32_f16(
                           false, pA, false, bV0, (short)0, acc0, false, false);
                acc1 = __builtin_amdgcn_wmma_f32_16x16x32_f16(
                           false, pA, false, bV1, (short)0, acc1, false, false);
                // softmax denominators: row-sum of P via WMMA against ones
                accL = __builtin_amdgcn_wmma_f32_16x16x32_f16(
                           false, pA, false, onesB, (short)0, accL, false, false);
            }

            // normalize and write back in place (f16)
#pragma unroll
            for (int r = 0; r < 8; ++r) {
                const float inv  = 1.f / accL[r];
                const int   rowi = mt * 16 + r + hi * 8;
                sQ[rowi * DIM + co + m]      = (_Float16)(acc0[r] * inv);
                sQ[rowi * DIM + co + 16 + m] = (_Float16)(acc1[r] * inv);
            }
        }
    }
    __syncthreads();

    // =================================================================
    // Phase 3: mean over the 6 replicas (linear -> hoist before Wp GEMM)
    //          Abar (64x128) written into sQ rows 0..63 (owner-exclusive)
    // =================================================================
    {
        const int r  = tid >> 2;           // 0..63
        const int c0 = (tid & 3) * 32;
#pragma unroll
        for (int ch = 0; ch < 4; ++ch) {
            float acc[8] = {};
#pragma unroll
            for (int n = 0; n < NREP; ++n) {
                v8h x = *(const v8h*)(sQ + (n * ROWS_OUT + r) * DIM + c0 + ch * 8);
#pragma unroll
                for (int j = 0; j < 8; ++j) acc[j] += (float)x[j];
            }
            v8h o;
#pragma unroll
            for (int j = 0; j < 8; ++j) o[j] = (_Float16)(acc[j] * (1.f / 6.f));
            *(v8h*)(sQ + r * DIM + c0 + ch * 8) = o;
        }
        // stage Wp swizzled (attention is done, scratch region is free again)
        stage_weight(Wp, sW, tid);
    }
    __syncthreads();

    // =================================================================
    // Phase 4: output projection 64x128x128 + store
    //   each wave owns one N-tile (nt == wave) and all 4 M-tiles:
    //   K-outermost, 4 independent accumulators, B loaded once per K-step
    // =================================================================
    {
        float* obase = out + (size_t)l * (ROWS_OUT * DIM);
        const int nt = wave;                       // 8 waves -> 8 N-tiles
        v8f cc[4];
#pragma unroll
        for (int mt = 0; mt < 4; ++mt) cc[mt] = (v8f){};
#pragma unroll
        for (int f = 0; f < 4; ++f) {
            const v16h B =
                *(const v16h*)(sW + ((nt * 8 + f * 2 + hi) << 8) + (m << 4));
#pragma unroll
            for (int mt = 0; mt < 4; ++mt) {
                const v16h A = loadA(sQ + (mt * 16 + m) * DIM + f * 32, hi);
                cc[mt] = __builtin_amdgcn_wmma_f32_16x16x32_f16(
                             false, A, false, B, (short)0, cc[mt], false, false);
            }
        }
        const float bb = bp[nt * 16 + m];
#pragma unroll
        for (int mt = 0; mt < 4; ++mt)
#pragma unroll
            for (int r = 0; r < 8; ++r) {
                const int rowi = mt * 16 + r + hi * 8;
                obase[rowi * DIM + nt * 16 + m] = cc[mt][r] + bb;
            }
    }
}

// =====================================================================
extern "C" void kernel_launch(void* const* d_in, const int* in_sizes, int n_in,
                              void* d_out, int out_size, void* d_ws, size_t ws_size,
                              hipStream_t stream) {
    (void)in_sizes; (void)n_in; (void)out_size; (void)d_ws; (void)ws_size;
    const float* q   = (const float*)d_in[0];
    const float* k   = (const float*)d_in[1];
    const float* v   = (const float*)d_in[2];
    const float* lqg = (const float*)d_in[3];
    const float* lqb = (const float*)d_in[4];
    const float* lkg = (const float*)d_in[5];
    const float* lkb = (const float*)d_in[6];
    const float* lvg = (const float*)d_in[7];
    const float* lvb = (const float*)d_in[8];
    const float* Wq  = (const float*)d_in[9];
    const float* bq  = (const float*)d_in[10];
    const float* Wk  = (const float*)d_in[11];
    const float* bk  = (const float*)d_in[12];
    const float* Wv  = (const float*)d_in[13];
    const float* bv  = (const float*)d_in[14];
    const float* Wp  = (const float*)d_in[15];
    const float* bp  = (const float*)d_in[16];
    float* out = (float*)d_out;

    const size_t shmem = (size_t)(3 * T_TOK * DIM + DIM * DIM) * sizeof(_Float16); // 320 KB
    hipFuncSetAttribute((const void*)xwin_attn_kernel,
                        hipFuncAttributeMaxDynamicSharedMemorySize, (int)shmem);
    xwin_attn_kernel<<<L_WIN, 256, shmem, stream>>>(
        q, k, v, lqg, lqb, lkg, lkb, lvg, lvb,
        Wq, bq, Wk, bk, Wv, bv, Wp, bp, out);
}
